// FusedMoE_8778913153198
// MI455X (gfx1250) — compile-verified, matrix operand order checked by reading, not run on hardware
//
#include <hip/hip_runtime.h>

// ---------------- problem constants ----------------
constexpr int kT = 1024;   // tokens
constexpr int kE = 8;      // experts
constexpr int kH = 1024;   // hidden
constexpr int kI = 768;    // intermediate
constexpr int kTopK = 2;
constexpr int kPad = 8;    // LDS row padding in halves (16B) to break bank conflicts
constexpr int kKC = 256;   // K-chunk (halves) staged in LDS (both GEMMs)
constexpr int kMT = 64;    // M rows per block (4 A-subtiles per B-fragment)

typedef __bf16 bf16;
typedef __attribute__((ext_vector_type(16))) __bf16 v16bf;
typedef __attribute__((ext_vector_type(8)))  __bf16 v8bf;
typedef __attribute__((ext_vector_type(4)))  __bf16 v4bf;
typedef __attribute__((ext_vector_type(8)))  float  v8f;
typedef __attribute__((ext_vector_type(4)))  float  v4f;

// ---------------- workspace layout (bytes) ----------------
constexpr size_t OFF_COUNTS = 0;                                           // kE ints
constexpr size_t OFF_TOK    = 256;                                         // kE*kT ints
constexpr size_t OFF_SLOT   = OFF_TOK  + sizeof(int)   * (size_t)kE * kT;
constexpr size_t OFF_WGT    = OFF_SLOT + sizeof(int)   * (size_t)kE * kT;
constexpr size_t OFF_W13    = (OFF_WGT + sizeof(float) * (size_t)kE * kT + 255) & ~(size_t)255;
constexpr size_t OFF_W2     = OFF_W13 + 2ull * kE * 2 * kI * kH;           // bf16
constexpr size_t OFF_ACT    = OFF_W2  + 2ull * kE * kH * kI;               // bf16
constexpr size_t OFF_PART   = OFF_ACT + 2ull * kE * kT * kI;               // fp32 [T*2][H]

__device__ __forceinline__ bf16 f2bf(float f) {
    unsigned u = __builtin_bit_cast(unsigned, f);
    unsigned r = (u + 0x7FFFu + ((u >> 16) & 1u)) >> 16;   // RNE
    return __builtin_bit_cast(bf16, (unsigned short)r);
}
__device__ __forceinline__ v16bf cat16(v8bf lo, v8bf hi) {
    return __builtin_shufflevector(lo, hi, 0,1,2,3,4,5,6,7,8,9,10,11,12,13,14,15);
}

// ---------------- routing ----------------
__global__ void moe_zero_counts(int* counts) {
    if (threadIdx.x < kE) counts[threadIdx.x] = 0;
}

__global__ void moe_route(const float* __restrict__ logits, int* __restrict__ counts,
                          int* __restrict__ tok, int* __restrict__ slot,
                          float* __restrict__ wgt) {
    int t = blockIdx.x * blockDim.x + threadIdx.x;
    if (t >= kT) return;
    float l[kE];
    v4f a = *(const v4f*)(logits + (size_t)t * kE);
    v4f b = *(const v4f*)(logits + (size_t)t * kE + 4);
#pragma unroll
    for (int i = 0; i < 4; ++i) { l[i] = a[i]; l[4 + i] = b[i]; }
    int i0 = 0; float b0 = l[0];
#pragma unroll
    for (int e = 1; e < kE; ++e) if (l[e] > b0) { b0 = l[e]; i0 = e; }
    int i1 = -1; float b1 = -3.4e38f;
#pragma unroll
    for (int e = 0; e < kE; ++e) if (e != i0 && l[e] > b1) { b1 = l[e]; i1 = e; }
    // softmax denominator cancels in top-k renormalization:
    float w0 = 1.f / (1.f + __expf(b1 - b0));
    float w1 = 1.f - w0;
    int r0 = atomicAdd(&counts[i0], 1);
    int row0 = i0 * kT + r0;
    tok[row0] = t; slot[row0] = 0; wgt[row0] = w0;
    int r1 = atomicAdd(&counts[i1], 1);
    int row1 = i1 * kT + r1;
    tok[row1] = t; slot[row1] = 1; wgt[row1] = w1;
}

// ---------------- dequant fp32 -> bf16 (scale folded once) ----------------
__global__ void moe_dequant(const float* __restrict__ w, const float* __restrict__ s,
                            bf16* __restrict__ o, int K, long total) {
    long idx = (blockIdx.x * (long)blockDim.x + threadIdx.x) * 4;
    if (idx >= total) return;
    long row = idx / K;
    int  k   = (int)(idx - row * K);
    float sc = s[row * (K >> 7) + (k >> 7)];
    v4f wv = *(const v4f*)(w + idx);
    v4bf ov;
#pragma unroll
    for (int i = 0; i < 4; ++i) ov[i] = f2bf(wv[i] * sc);
    *(v4bf*)(o + idx) = ov;
}

// ---------------- GEMM1: act = silu(x@Wg^T) * (x@Wu^T), gathered rows ----------------
// grid: (kI/64, kT/kMT, kE), block 128 (4 waves).
// Wave owns 16 N-cols (gate+up pair) and M=64 rows: 8 WMMAs per 2 B-fragments.
__global__ void moe_gemm1(const float* __restrict__ x, const bf16* __restrict__ w13,
                          const int* __restrict__ counts, const int* __restrict__ tok,
                          bf16* __restrict__ act) {
    const int e   = blockIdx.z;
    const int m0  = blockIdx.y * kMT;
    const int cnt = counts[e];
    if (m0 >= cnt) return;

    __shared__ bf16 As[kMT * (kKC + kPad)];   // 33,792 B
    const int tid  = threadIdx.x;
    const int wave = tid >> 5, lane = tid & 31;
    const int ncol = blockIdx.x * 64 + wave * 16 + (lane & 15);
    const bf16* wg = w13 + ((size_t)e * 2 * kI + ncol) * kH;  // gate row (W^T load == A layout)
    const bf16* wu = wg + (size_t)kI * kH;                    // up row
    const int m   = lane & 15;
    const int khi = (lane >> 4) * 8;  // lanes 16-31 hold K 8-15 / 24-31

    v8f cg[4], cu[4];
#pragma unroll
    for (int r = 0; r < 4; ++r) {
        cg[r] = (v8f){0.f,0.f,0.f,0.f,0.f,0.f,0.f,0.f};
        cu[r] = (v8f){0.f,0.f,0.f,0.f,0.f,0.f,0.f,0.f};
    }

    for (int kc = 0; kc < kH; kc += kKC) {
        if (kc != 0) __syncthreads();
        // stage kMT gathered token rows of x[:, kc:kc+kKC) as bf16 into LDS
        for (int c = tid; c < kMT * (kKC / 4); c += blockDim.x) {
            int j    = c / (kKC / 4);
            int colc = (c % (kKC / 4)) * 4;
            v4bf v;
#pragma unroll
            for (int i = 0; i < 4; ++i) v[i] = f2bf(0.f);
            if (m0 + j < cnt) {
                int t = tok[e * kT + m0 + j];
                v4f xv = *(const v4f*)(x + (size_t)t * kH + kc + colc);
#pragma unroll
                for (int i = 0; i < 4; ++i) v[i] = f2bf(xv[i]);
            }
            *(v4bf*)(&As[j * (kKC + kPad) + colc]) = v;
        }
        __syncthreads();

        for (int kk = 0; kk < kKC; kk += 32) {
            const int k0 = kk + khi;        // LDS-chunk-relative k
            const int kg = kc + k0;         // global k
            __builtin_prefetch(wg + kc + kk + 256, 0, 1);
            __builtin_prefetch(wu + kc + kk + 256, 0, 1);
            v16bf bg = cat16(*(const v8bf*)(wg + kg), *(const v8bf*)(wg + kg + 16));
            v16bf bu = cat16(*(const v8bf*)(wu + kg), *(const v8bf*)(wu + kg + 16));
#pragma unroll
            for (int r = 0; r < 4; ++r) {
                const int mrow = m + 16 * r;
                v16bf af = cat16(*(const v8bf*)(&As[mrow * (kKC + kPad) + k0]),
                                 *(const v8bf*)(&As[mrow * (kKC + kPad) + k0 + 16]));
                cg[r] = __builtin_amdgcn_wmma_f32_16x16x32_bf16(false, af, false, bg, (short)0, cg[r], false, false);
                cu[r] = __builtin_amdgcn_wmma_f32_16x16x32_bf16(false, af, false, bu, (short)0, cu[r], false, false);
            }
        }
    }

    const int rbase = (lane >> 4) * 8;  // C layout: VGPR v -> M = v (+8 for hi lanes)
#pragma unroll
    for (int r = 0; r < 4; ++r) {
#pragma unroll
        for (int v = 0; v < 8; ++v) {
            int ml = rbase + v + 16 * r;
            if (m0 + ml < cnt) {
                float g = cg[r][v], u = cu[r][v];
                float a = (g / (1.f + __expf(-g))) * u;   // SiLU(g) * u
                act[((size_t)(e * kT + m0 + ml)) * kI + ncol] = f2bf(a);
            }
        }
    }
}

// ---------------- GEMM2: part[t,slot] = wgt * (act @ W2^T) ----------------
// grid: (kH/64, kT/kMT, kE), block 128; M=64 per wave (4 WMMAs per B-fragment)
__global__ void moe_gemm2(const bf16* __restrict__ act, const bf16* __restrict__ w2,
                          const int* __restrict__ counts, const int* __restrict__ tok,
                          const int* __restrict__ slot, const float* __restrict__ wgt,
                          float* __restrict__ part) {
    const int e   = blockIdx.z;
    const int m0  = blockIdx.y * kMT;
    const int cnt = counts[e];
    if (m0 >= cnt) return;

    __shared__ bf16 As[kMT * (kKC + kPad)];   // 33,792 B
    const int tid  = threadIdx.x;
    const int wave = tid >> 5, lane = tid & 31;
    const int hcol = blockIdx.x * 64 + wave * 16 + (lane & 15);
    const bf16* wr = w2 + ((size_t)e * kH + hcol) * kI;
    const int m   = lane & 15;
    const int khi = (lane >> 4) * 8;

    v8f cacc[4];
#pragma unroll
    for (int r = 0; r < 4; ++r) cacc[r] = (v8f){0.f,0.f,0.f,0.f,0.f,0.f,0.f,0.f};

    for (int kc = 0; kc < kI; kc += kKC) {
        if (kc != 0) __syncthreads();
        // stage kMT rows of act[:, kc:kc+kKC) into LDS (already bf16)
        for (int c = tid; c < kMT * (kKC / 8); c += blockDim.x) {
            int j    = c / (kKC / 8);
            int colc = (c % (kKC / 8)) * 8;
            v8bf v;
            if (m0 + j < cnt) {
                v = *(const v8bf*)(act + ((size_t)(e * kT + m0 + j)) * kI + kc + colc);
            } else {
#pragma unroll
                for (int i = 0; i < 8; ++i) v[i] = f2bf(0.f);
            }
            *(v8bf*)(&As[j * (kKC + kPad) + colc]) = v;
        }
        __syncthreads();

        for (int kk = 0; kk < kKC; kk += 32) {
            const int k0 = kk + khi;
            const int kg = kc + k0;
            __builtin_prefetch(wr + kc + kk + 256, 0, 1);
            v16bf bfr = cat16(*(const v8bf*)(wr + kg), *(const v8bf*)(wr + kg + 16));
#pragma unroll
            for (int r = 0; r < 4; ++r) {
                const int mrow = m + 16 * r;
                v16bf af = cat16(*(const v8bf*)(&As[mrow * (kKC + kPad) + k0]),
                                 *(const v8bf*)(&As[mrow * (kKC + kPad) + k0 + 16]));
                cacc[r] = __builtin_amdgcn_wmma_f32_16x16x32_bf16(false, af, false, bfr, (short)0, cacc[r], false, false);
            }
        }
    }

    const int rbase = (lane >> 4) * 8;
#pragma unroll
    for (int r = 0; r < 4; ++r) {
#pragma unroll
        for (int v = 0; v < 8; ++v) {
            int ml = rbase + v + 16 * r;
            int ri = m0 + ml;
            if (ri < cnt) {
                int rr = e * kT + ri;
                int t  = tok[rr];
                part[((size_t)(t * kTopK + slot[rr])) * kH + hcol] = wgt[rr] * cacc[r][v];
            }
        }
    }
}

// ---------------- final combine of the two expert slots ----------------
__global__ void moe_combine(const float* __restrict__ part, float* __restrict__ out) {
    long i = (blockIdx.x * (long)blockDim.x + threadIdx.x) * 4;
    long t = i / kH;
    int  h = (int)(i - t * kH);
    v4f a = *(const v4f*)(part + (t * 2 + 0) * (long)kH + h);
    v4f b = *(const v4f*)(part + (t * 2 + 1) * (long)kH + h);
    *(v4f*)(out + i) = a + b;
}

// ---------------- launcher ----------------
extern "C" void kernel_launch(void* const* d_in, const int* in_sizes, int n_in,
                              void* d_out, int out_size, void* d_ws, size_t ws_size,
                              hipStream_t stream) {
    const float* x   = (const float*)d_in[0];
    const float* rl  = (const float*)d_in[1];
    const float* w13 = (const float*)d_in[2];
    const float* s13 = (const float*)d_in[3];
    const float* w2  = (const float*)d_in[4];
    const float* s2  = (const float*)d_in[5];
    float* out = (float*)d_out;

    char* ws = (char*)d_ws;
    int*   counts = (int*)  (ws + OFF_COUNTS);
    int*   tok    = (int*)  (ws + OFF_TOK);
    int*   slot   = (int*)  (ws + OFF_SLOT);
    float* wgt    = (float*)(ws + OFF_WGT);
    bf16*  w13b   = (bf16*) (ws + OFF_W13);
    bf16*  w2b    = (bf16*) (ws + OFF_W2);
    bf16*  actb   = (bf16*) (ws + OFF_ACT);
    float* part   = (float*)(ws + OFF_PART);

    moe_zero_counts<<<1, 32, 0, stream>>>(counts);
    moe_route<<<kT / 256, 256, 0, stream>>>(rl, counts, tok, slot, wgt);

    long n13 = (long)kE * 2 * kI * kH;
    long n2  = (long)kE * kH * kI;
    moe_dequant<<<(unsigned)((n13 / 4 + 255) / 256), 256, 0, stream>>>(w13, s13, w13b, kH, n13);
    moe_dequant<<<(unsigned)((n2  / 4 + 255) / 256), 256, 0, stream>>>(w2,  s2,  w2b,  kI, n2);

    moe_gemm1<<<dim3(kI / 64, kT / kMT, kE), 128, 0, stream>>>(x, w13b, counts, tok, actb);
    moe_gemm2<<<dim3(kH / 64, kT / kMT, kE), 128, 0, stream>>>(actb, w2b, counts, tok, slot, wgt, part);
    moe_combine<<<(unsigned)(((long)kT * kH / 4) / 256), 256, 0, stream>>>(part, out);
}
